// ChannelAttentionModule_5523327942725
// MI455X (gfx1250) — matrix operand order repaired and checked by reading, not compiled, not run on hardware
//
#include <hip/hip_runtime.h>
#include <hip/hip_bf16.h>
#include <cstdint>
#include <cstddef>

// ---------------------------------------------------------------------------
// Channel co-attention for MI455X (gfx1250, wave32, WMMA bf16 path)
//   x1,x2: [32, 512, 64, 64] f32
// ---------------------------------------------------------------------------

typedef __bf16 bf16_t;
typedef __attribute__((ext_vector_type(16))) __bf16 v16bf;
typedef __attribute__((ext_vector_type(8)))  __bf16 v8bf;
typedef __attribute__((ext_vector_type(8)))  float  v8f;

#define N_BATCH 32
#define C_DIM   512
#define HW_DIM  4096
#define EPS_NRM 1e-12f

static constexpr size_t NC   = (size_t)N_BATCH * C_DIM;           // 16384 rows
static constexpr size_t NCHW = (size_t)N_BATCH * C_DIM * HW_DIM;  // 67,108,864
static constexpr size_t NCC  = (size_t)N_BATCH * C_DIM * C_DIM;   // 8,388,608

// ---------------- WMMA fragment helpers (contiguous K order for A and B) -----
// A 16x32 bf16: lane(0-15)=row M, laneHi selects K-half; elements
//   e=0..15 -> K = 16*laneHi + e
__device__ __forceinline__ v16bf load_fragA(const bf16_t* rowBase, int k0, int laneHi) {
    return *(const v16bf*)(rowBase + k0 + laneHi * 16);
}

// B 32x16 bf16: lane(0-15)=col N, per-lane 16 consecutive K at 16*laneHi.
__device__ __forceinline__ v16bf load_fragB(const bf16_t* colRowBase, int k0, int laneHi) {
    return *(const v16bf*)(colRowBase + k0 + laneHi * 16);
}

__device__ __forceinline__ v8f wmma_bf16(v16bf a, v16bf b, v8f c) {
    // D = A x B + C  (f32 accum), emits v_wmma_f32_16x16x32_bf16
    return __builtin_amdgcn_wmma_f32_16x16x32_bf16(
        /*neg_a=*/false, a, /*neg_b=*/false, b,
        /*c_mod=*/(short)0, c, /*reuse_a=*/false, /*reuse_b=*/false);
}

// ---------------- Kernel 0: norms + bf16 cast + bf16 transpose -------------
// One 256-thread block per (batch, channel) row of 4096 floats.
__global__ void __launch_bounds__(256)
prep_kernel(const float* __restrict__ x, bf16_t* __restrict__ xb,
            bf16_t* __restrict__ xtb, float* __restrict__ nrm) {
    const int row = blockIdx.x;              // b*C + ch
    const int b   = row / C_DIM;
    const int ch  = row - b * C_DIM;
    const int tid = threadIdx.x;

    const float* src  = x   + (size_t)row * HW_DIM;
    bf16_t*      dst  = xb  + (size_t)row * HW_DIM;
    bf16_t*      dstT = xtb + (size_t)b * HW_DIM * C_DIM + ch;

    float ss = 0.f;
    for (int s = tid; s < HW_DIM; s += 256) {
        const float v = src[s];
        ss += v * v;
        const bf16_t bv = (bf16_t)v;
        dst[s] = bv;
        dstT[(size_t)s * C_DIM] = bv;
    }
    __shared__ float red[256];
    red[tid] = ss;
    __syncthreads();
    for (int off = 128; off > 0; off >>= 1) {
        if (tid < off) red[tid] += red[tid + off];
        __syncthreads();
    }
    if (tid == 0) nrm[row] = sqrtf(red[0]);
}

// ---------------- Kernel 1: att = (x1b . x2b^T) / (n1 n2) ------------------
// Block = 256 thr (8 waves), tile 128x128; wave tile 32x64 (2x4 WMMA accums).
__global__ void __launch_bounds__(256)
gemm1_kernel(const bf16_t* __restrict__ x1b, const bf16_t* __restrict__ x2b,
             const float* __restrict__ nrm1, const float* __restrict__ nrm2,
             float* __restrict__ att) {
    const int b      = blockIdx.z;
    const int tid    = threadIdx.x;
    const int lane   = tid & 31;
    const int w      = tid >> 5;
    const int laneLo = lane & 15;
    const int laneHi = lane >> 4;
    const int m0 = blockIdx.y * 128 + (w & 3) * 32;   // rows (c1)
    const int n0 = blockIdx.x * 128 + (w >> 2) * 64;  // cols (c2)

    const bf16_t* Abase = x1b + ((size_t)b * C_DIM + m0 + laneLo) * HW_DIM;
    const bf16_t* Bbase = x2b + ((size_t)b * C_DIM + n0 + laneLo) * HW_DIM;

    v8f acc[2][4] = {};
    for (int k0 = 0; k0 < HW_DIM; k0 += 32) {
        v16bf a[2], bb[4];
        #pragma unroll
        for (int i = 0; i < 2; ++i)
            a[i] = load_fragA(Abase + (size_t)i * 16 * HW_DIM, k0, laneHi);
        #pragma unroll
        for (int j = 0; j < 4; ++j)
            bb[j] = load_fragB(Bbase + (size_t)j * 16 * HW_DIM, k0, laneHi);
        #pragma unroll
        for (int i = 0; i < 2; ++i)
            #pragma unroll
            for (int j = 0; j < 4; ++j)
                acc[i][j] = wmma_bf16(a[i], bb[j], acc[i][j]);
    }

    // epilogue: fold in the L2-normalization factors, store f32 attention
    const float* n1 = nrm1 + (size_t)b * C_DIM;
    const float* n2 = nrm2 + (size_t)b * C_DIM;
    #pragma unroll
    for (int j = 0; j < 4; ++j) {
        const int   col = n0 + j * 16 + laneLo;
        const float s2  = 1.f / fmaxf(n2[col], EPS_NRM);
        #pragma unroll
        for (int i = 0; i < 2; ++i) {
            #pragma unroll
            for (int r = 0; r < 8; ++r) {
                const int   rowi = m0 + i * 16 + laneHi * 8 + r;  // C/D layout
                const float s1   = 1.f / fmaxf(n1[rowi], EPS_NRM);
                att[((size_t)b * C_DIM + rowi) * C_DIM + col] = acc[i][j][r] * s1 * s2;
            }
        }
    }
}

// ---------------- Kernel 2: row softmax -> bf16 attn1 ----------------------
__global__ void __launch_bounds__(256)
softmax_row_kernel(const float* __restrict__ att, bf16_t* __restrict__ out) {
    const size_t base = (size_t)blockIdx.x * C_DIM;   // (b*C + i) row
    const int tid = threadIdx.x;
    const float v0 = att[base + tid];
    const float v1 = att[base + tid + 256];

    __shared__ float red[256];
    red[tid] = fmaxf(v0, v1);
    __syncthreads();
    for (int off = 128; off > 0; off >>= 1) {
        if (tid < off) red[tid] = fmaxf(red[tid], red[tid + off]);
        __syncthreads();
    }
    const float mx = red[0];
    __syncthreads();

    const float e0 = __expf(v0 - mx);
    const float e1 = __expf(v1 - mx);
    red[tid] = e0 + e1;
    __syncthreads();
    for (int off = 128; off > 0; off >>= 1) {
        if (tid < off) red[tid] += red[tid + off];
        __syncthreads();
    }
    const float inv = 1.f / red[0];
    out[base + tid]       = (bf16_t)(e0 * inv);
    out[base + tid + 256] = (bf16_t)(e1 * inv);
}

// ---------------- Kernel 3: column softmax, stored transposed --------------
// out[b][j][i] = softmax_i(att[b][i][j])  (row-major in i => K-contiguous A)
__global__ void __launch_bounds__(256)
softmax_col_kernel(const float* __restrict__ att, bf16_t* __restrict__ out) {
    const int bj  = blockIdx.x;                 // b*C + j
    const int b   = bj / C_DIM;
    const int j   = bj - b * C_DIM;
    const int tid = threadIdx.x;

    const float* colBase = att + (size_t)b * C_DIM * C_DIM + j;
    const float v0 = colBase[(size_t)tid * C_DIM];
    const float v1 = colBase[(size_t)(tid + 256) * C_DIM];

    __shared__ float red[256];
    red[tid] = fmaxf(v0, v1);
    __syncthreads();
    for (int off = 128; off > 0; off >>= 1) {
        if (tid < off) red[tid] = fmaxf(red[tid], red[tid + off]);
        __syncthreads();
    }
    const float mx = red[0];
    __syncthreads();

    const float e0 = __expf(v0 - mx);
    const float e1 = __expf(v1 - mx);
    red[tid] = e0 + e1;
    __syncthreads();
    for (int off = 128; off > 0; off >>= 1) {
        if (tid < off) red[tid] += red[tid + off];
        __syncthreads();
    }
    const float inv = 1.f / red[0];
    bf16_t* o = out + (size_t)bj * C_DIM;
    o[tid]       = (bf16_t)(e0 * inv);
    o[tid + 256] = (bf16_t)(e1 * inv);
}

// ---------------- Kernel 4/5: out = scale*(attn @ X) + residual ------------
// A: bf16 [n][C][C] (K contiguous). B from transposed bf16 copy [n][HW][C]
// so each fragment load is K-contiguous. Fused residual + alpha/beta scale.
__global__ void __launch_bounds__(256)
gemm2_kernel(const bf16_t* __restrict__ attn, const bf16_t* __restrict__ xtb,
             const float* __restrict__ resid, const float* __restrict__ scale_p,
             float* __restrict__ out) {
    const int b      = blockIdx.z;
    const int tid    = threadIdx.x;
    const int lane   = tid & 31;
    const int w      = tid >> 5;
    const int laneLo = lane & 15;
    const int laneHi = lane >> 4;
    const int m0 = blockIdx.y * 128 + (w & 3) * 32;   // rows (channels)
    const int n0 = blockIdx.x * 128 + (w >> 2) * 64;  // cols (spatial)
    const float scale = scale_p[0];

    const bf16_t* Abase = attn + ((size_t)b * C_DIM + m0 + laneLo) * C_DIM;
    const bf16_t* Bbase = xtb + ((size_t)b * HW_DIM + n0 + laneLo) * C_DIM;

    v8f acc[2][4] = {};
    for (int k0 = 0; k0 < C_DIM; k0 += 32) {
        v16bf a[2], bb[4];
        #pragma unroll
        for (int i = 0; i < 2; ++i)
            a[i] = load_fragA(Abase + (size_t)i * 16 * C_DIM, k0, laneHi);
        #pragma unroll
        for (int j = 0; j < 4; ++j)
            bb[j] = load_fragB(Bbase + (size_t)j * 16 * C_DIM, k0, laneHi);
        #pragma unroll
        for (int i = 0; i < 2; ++i)
            #pragma unroll
            for (int j = 0; j < 4; ++j)
                acc[i][j] = wmma_bf16(a[i], bb[j], acc[i][j]);
    }

    #pragma unroll
    for (int j = 0; j < 4; ++j) {
        const int col = n0 + j * 16 + laneLo;
        #pragma unroll
        for (int i = 0; i < 2; ++i) {
            #pragma unroll
            for (int r = 0; r < 8; ++r) {
                const int    rowi = m0 + i * 16 + laneHi * 8 + r;
                const size_t idx  = ((size_t)b * C_DIM + rowi) * HW_DIM + col;
                out[idx] = scale * acc[i][j][r] + resid[idx];
            }
        }
    }
}

// ---------------------------------------------------------------------------
extern "C" void kernel_launch(void* const* d_in, const int* in_sizes, int n_in,
                              void* d_out, int out_size, void* d_ws, size_t ws_size,
                              hipStream_t stream) {
    const float* x1    = (const float*)d_in[0];
    const float* x2    = (const float*)d_in[1];
    const float* alpha = (const float*)d_in[2];
    const float* beta  = (const float*)d_in[3];

    float* out1 = (float*)d_out;
    float* out2 = out1 + NCHW;

    // workspace carve-up (all offsets 256B aligned)
    char*   ws   = (char*)d_ws;
    size_t  off  = 0;
    bf16_t* x1b  = (bf16_t*)(ws + off); off += NCHW * sizeof(bf16_t);
    bf16_t* x2b  = (bf16_t*)(ws + off); off += NCHW * sizeof(bf16_t);
    bf16_t* x1tb = (bf16_t*)(ws + off); off += NCHW * sizeof(bf16_t);
    bf16_t* x2tb = (bf16_t*)(ws + off); off += NCHW * sizeof(bf16_t);
    float*  att  = (float*)(ws + off);  off += NCC * sizeof(float);
    bf16_t* a1b  = (bf16_t*)(ws + off); off += NCC * sizeof(bf16_t);
    bf16_t* a2b  = (bf16_t*)(ws + off); off += NCC * sizeof(bf16_t);
    float*  n1   = (float*)(ws + off);  off += NC * sizeof(float);
    float*  n2   = (float*)(ws + off);  off += NC * sizeof(float);
    (void)ws_size; (void)in_sizes; (void)n_in; (void)out_size;

    // 1) norms + bf16 casts + transposed copies
    prep_kernel<<<dim3((unsigned)NC), dim3(256), 0, stream>>>(x1, x1b, x1tb, n1);
    prep_kernel<<<dim3((unsigned)NC), dim3(256), 0, stream>>>(x2, x2b, x2tb, n2);

    // 2) attention Gram matrix with fused normalization (WMMA bf16)
    gemm1_kernel<<<dim3(4, 4, N_BATCH), dim3(256), 0, stream>>>(x1b, x2b, n1, n2, att);

    // 3) softmaxes (row -> attn1; column stored transposed -> attn2^T)
    softmax_row_kernel<<<dim3((unsigned)NC), dim3(256), 0, stream>>>(att, a1b);
    softmax_col_kernel<<<dim3((unsigned)NC), dim3(256), 0, stream>>>(att, a2b);

    // 4) apply attention + residual (WMMA bf16), straight into d_out
    gemm2_kernel<<<dim3(32, 4, N_BATCH), dim3(256), 0, stream>>>(a1b, x2tb, x1, alpha, out1);
    gemm2_kernel<<<dim3(32, 4, N_BATCH), dim3(256), 0, stream>>>(a2b, x1tb, x2, beta, out2);
}
